// HyenaEncoderBlock_66941360275914
// MI455X (gfx1250) — matrix operand order repaired and checked by reading, not compiled, not run on hardware
//
#include <hip/hip_runtime.h>
#include <stdint.h>

// ---------------------------------------------------------------------------
// HyenaEncoderBlock for MI455X (gfx1250), wave32 + WMMA bf16x2-split GEMMs.
//   B=4, L_IN=2048, L_OUT=1024, D=1024, D_POS=256, DH=1280
// Pipeline:
//   z1 = z @ W_in + b_in                      (WMMA GEMM)
//   z1 = LN(z1, g_ln, beta_ln)                (in-place)
//   t1 = silu(pe @ Wp1 + bp1)                 (WMMA GEMM)
//   h  = t1 @ Wp2 + bp2                       (WMMA GEMM)
//   h *= exp(-l * exp(a))                     (window, in-place)
//   xln = LN(x, g1, be1)
//   mid = silu(xln @ W1a + b1a)               (WMMA GEMM)
//   xr  = mid @ W1b + b1b + x                 (WMMA GEMM + residual)
//   y1[b,j,d] = xr[b,j,d] * sum_k h[k,d]*z1[b,2j-k,d]   (direct causal conv,
//                                              replaces rfft/irfft exactly)
//   xln = LN(y1, g2, be2)
//   mid = silu(xln @ W2a + b2a)               (WMMA GEMM)
//   out = mid @ W2b + b2b + y1                (WMMA GEMM + residual)
// ---------------------------------------------------------------------------

typedef __attribute__((ext_vector_type(16))) __bf16    v16bf;
typedef __attribute__((ext_vector_type(8)))  float     v8f;
typedef __attribute__((ext_vector_type(8)))  uint32_t  v8u;

#define L_IN_   2048
#define L_OUT_  1024
#define DMODEL  1024
#define BATCH_  4
#define EPS_    1e-5f

// Split a float pair into packed bf16 hi and bf16 lo (residual) dwords.
// Element 0 -> bits [15:0] (even K), element 1 -> bits [31:16] (odd K).
static __device__ __forceinline__ uint32_t pack_hilo(float a, float b, uint32_t& lo_out) {
  __bf16 ah = (__bf16)a;
  __bf16 bh = (__bf16)b;
  __bf16 al = (__bf16)(a - (float)ah);
  __bf16 bl = (__bf16)(b - (float)bh);
  union { __bf16 h[2]; uint32_t u; } uh, ul;
  uh.h[0] = ah; uh.h[1] = bh;
  ul.h[0] = al; ul.h[1] = bl;
  lo_out = ul.u;
  return uh.u;
}

static __device__ __forceinline__ v16bf load_frag(const uint32_t* p) {
  return __builtin_bit_cast(v16bf, *reinterpret_cast<const v8u*>(p));
}

static __device__ __forceinline__ v8f wmma_bf16(v16bf a, v16bf b, v8f c) {
  return __builtin_amdgcn_wmma_f32_16x16x32_bf16(false, a, false, b, (short)0, c,
                                                 false, false);
}

// Fast silu: v * rcp(1 + exp(-v)); v_rcp_f32 is ~1 ulp, far below the
// bf16x2 GEMM noise floor, and avoids the 14-op IEEE division sequence.
static __device__ __forceinline__ float fast_silu(float v) {
  return v * __builtin_amdgcn_rcpf(1.0f + __expf(-v));
}

// ---------------------------------------------------------------------------
// GEMM: out[M,N] = act(A[M,K] @ W[K,N] + bias[N]) (+ res[M,N])
// Block tile 128x64, 256 threads (8 waves), wave tile 32x32 (2x2 frags).
// K stepped by 32; A/B panels staged in LDS pre-swizzled into WMMA fragment
// layout, split into bf16 hi/lo (3 WMMAs/frag-step recover ~fp32 accuracy).
// Double-buffered LDS: global loads for step k+1 issue before the WMMA block
// of step k; one barrier per step.
// ---------------------------------------------------------------------------
template <bool SILU, bool RES>
__global__ __launch_bounds__(256) void gemm_bf16x3_k(
    const float* __restrict__ A, const float* __restrict__ W,
    const float* __restrict__ bias, const float* __restrict__ res,
    float* __restrict__ out, int M, int N, int K)
{
  __shared__ __align__(32) uint32_t sAh[2][8 * 32 * 8];
  __shared__ __align__(32) uint32_t sAl[2][8 * 32 * 8];
  __shared__ __align__(32) uint32_t sBh[2][4 * 32 * 8];
  __shared__ __align__(32) uint32_t sBl[2][4 * 32 * 8];

  const int tid   = threadIdx.x;
  const int lane  = tid & 31;
  const int wave  = tid >> 5;
  const int waveM = wave >> 1;          // 0..3 -> rows 32*waveM
  const int waveN = wave & 1;           // 0..1 -> cols 32*waveN
  const int m0    = blockIdx.y * 128;
  const int n0    = blockIdx.x * 64;
  const int kp    = tid >> 4;           // B stage: k pair 0..15
  const int nq    = tid & 15;           // B stage: n quad 0..15

  v8f c00 = {}, c01 = {}, c10 = {}, c11 = {};

  float4 ra[4], rb0, rb1;               // staged global data for next step

  auto g_load = [&](int k0) {
#pragma unroll
    for (int s = 0; s < 4; ++s) {
      int q   = tid + 256 * s;          // 0..1023
      int row = q >> 3;                 // 0..127
      int kq  = (q & 7) * 4;            // 0,4,...,28
      ra[s] = *reinterpret_cast<const float4*>(
          &A[(size_t)(m0 + row) * K + k0 + kq]);
    }
    const float* Wr = &W[(size_t)(k0 + 2 * kp) * N + n0 + nq * 4];
    rb0 = *reinterpret_cast<const float4*>(Wr);
    rb1 = *reinterpret_cast<const float4*>(Wr + N);
  };

  auto lds_store = [&](int buf) {
#pragma unroll
    for (int s = 0; s < 4; ++s) {
      int q   = tid + 256 * s;
      int row = q >> 3;
      int kq  = (q & 7) * 4;
      float e[4] = {ra[s].x, ra[s].y, ra[s].z, ra[s].w};
      int fm = row >> 4;
      int mr = row & 15;
#pragma unroll
      for (int j = 0; j < 2; ++j) {
        int kk = kq + 2 * j;
        // 16-bit A 16x32 layout: lanes0-15 K{0-7,16-23}, lanes16-31 K{8-15,24-31}
        int ln  = mr + (((kk >> 3) & 1) ? 16 : 0);
        int v   = ((kk & 16) ? 4 : 0) + ((kk & 7) >> 1);
        uint32_t lo, hi = pack_hilo(e[2 * j], e[2 * j + 1], lo);
        int off = (fm * 32 + ln) * 8 + v;
        sAh[buf][off] = hi;
        sAl[buf][off] = lo;
      }
    }
    float e0[4] = {rb0.x, rb0.y, rb0.z, rb0.w};
    float e1[4] = {rb1.x, rb1.y, rb1.z, rb1.w};
#pragma unroll
    for (int i2 = 0; i2 < 4; ++i2) {
      int n  = nq * 4 + i2;
      int fn = n >> 4;
      // 16-bit B 32x16 layout: lanes0-15 K{0-7,16-23}, lanes16-31 K{8-15,24-31}
      int ln = (n & 15) + (((kp >> 2) & 1) ? 16 : 0);
      int v  = ((kp & 8) ? 4 : 0) + (kp & 3);
      uint32_t lo, hi = pack_hilo(e0[i2], e1[i2], lo);
      int off = (fn * 32 + ln) * 8 + v;
      sBh[buf][off] = hi;
      sBl[buf][off] = lo;
    }
  };

  auto compute = [&](int buf) {
    const int a0 = ((waveM * 2 + 0) * 32 + lane) * 8;
    const int a1 = ((waveM * 2 + 1) * 32 + lane) * 8;
    const int b0 = ((waveN * 2 + 0) * 32 + lane) * 8;
    const int b1 = ((waveN * 2 + 1) * 32 + lane) * 8;
    v16bf ah0 = load_frag(&sAh[buf][a0]), al0 = load_frag(&sAl[buf][a0]);
    v16bf ah1 = load_frag(&sAh[buf][a1]), al1 = load_frag(&sAl[buf][a1]);
    v16bf bh0 = load_frag(&sBh[buf][b0]), bl0 = load_frag(&sBl[buf][b0]);
    v16bf bh1 = load_frag(&sBh[buf][b1]), bl1 = load_frag(&sBl[buf][b1]);

    c00 = wmma_bf16(ah0, bh0, c00);
    c00 = wmma_bf16(ah0, bl0, c00);
    c00 = wmma_bf16(al0, bh0, c00);

    c01 = wmma_bf16(ah0, bh1, c01);
    c01 = wmma_bf16(ah0, bl1, c01);
    c01 = wmma_bf16(al0, bh1, c01);

    c10 = wmma_bf16(ah1, bh0, c10);
    c10 = wmma_bf16(ah1, bl0, c10);
    c10 = wmma_bf16(al1, bh0, c10);

    c11 = wmma_bf16(ah1, bh1, c11);
    c11 = wmma_bf16(ah1, bl1, c11);
    c11 = wmma_bf16(al1, bh1, c11);
  };

  // ---- double-buffered pipeline ----
  const int nsteps = K >> 5;
  g_load(0);
  lds_store(0);
  __syncthreads();
  for (int i = 0; i < nsteps; ++i) {
    const int buf = i & 1;
    const bool has_next = (i + 1) < nsteps;
    if (has_next) g_load((i + 1) << 5);   // HBM/L2 latency hidden behind WMMAs
    compute(buf);
    if (has_next) lds_store(buf ^ 1);
    __syncthreads();
  }

  // ---- epilogue: C/D 16x16 layout: n = lane&15, m = r + 8*(lane>=16) ----
  const int nn = lane & 15;
  const int mb = (lane >> 4) * 8;
  const v8f* cs[2][2] = {{&c00, &c01}, {&c10, &c11}};
#pragma unroll
  for (int fm = 0; fm < 2; ++fm) {
#pragma unroll
    for (int fn = 0; fn < 2; ++fn) {
      int col  = n0 + (waveN * 2 + fn) * 16 + nn;
      float bv = bias[col];
      const v8f& cc = *cs[fm][fn];
#pragma unroll
      for (int r = 0; r < 8; ++r) {
        int row = m0 + (waveM * 2 + fm) * 16 + mb + r;
        float v = cc[r] + bv;
        if (SILU) v = fast_silu(v);
        if (RES)  v += res[(size_t)row * N + col];
        out[(size_t)row * N + col] = v;
      }
    }
  }
}

// ---------------------------------------------------------------------------
// LayerNorm over last dim (=1024). One 256-thread block per row, float4 I/O.
// ---------------------------------------------------------------------------
__global__ __launch_bounds__(256) void layernorm_k(
    const float* __restrict__ in, float* __restrict__ out,
    const float* __restrict__ g, const float* __restrict__ b)
{
  __shared__ float s1[256];
  __shared__ float s2[256];
  __shared__ float mv[2];
  const int tid = threadIdx.x;
  const size_t row = blockIdx.x;
  float4 v = reinterpret_cast<const float4*>(in + row * DMODEL)[tid];
  float sum = v.x + v.y + v.z + v.w;
  float sq  = v.x * v.x + v.y * v.y + v.z * v.z + v.w * v.w;
  s1[tid] = sum; s2[tid] = sq;
  __syncthreads();
  for (int off = 128; off > 0; off >>= 1) {
    if (tid < off) { s1[tid] += s1[tid + off]; s2[tid] += s2[tid + off]; }
    __syncthreads();
  }
  if (tid == 0) {
    float m   = s1[0] * (1.0f / DMODEL);
    float var = s2[0] * (1.0f / DMODEL) - m * m;
    mv[0] = m; mv[1] = rsqrtf(var + EPS_);
  }
  __syncthreads();
  const float m = mv[0], inv = mv[1];
  float4 gg = reinterpret_cast<const float4*>(g)[tid];
  float4 bb = reinterpret_cast<const float4*>(b)[tid];
  float4 o;
  o.x = (v.x - m) * inv * gg.x + bb.x;
  o.y = (v.y - m) * inv * gg.y + bb.y;
  o.z = (v.z - m) * inv * gg.z + bb.z;
  o.w = (v.w - m) * inv * gg.w + bb.w;
  reinterpret_cast<float4*>(out + row * DMODEL)[tid] = o;
}

// ---------------------------------------------------------------------------
// In-place window: h[l,d] *= exp(-l * exp(a[0]))
// ---------------------------------------------------------------------------
__global__ __launch_bounds__(256) void window_k(float* __restrict__ h,
                                                const float* __restrict__ a)
{
  int idx = blockIdx.x * 256 + threadIdx.x;     // < L_IN * D
  int l = idx >> 10;
  float decay = expf(a[0]);
  h[idx] *= expf(-(float)l * decay);
}

// ---------------------------------------------------------------------------
// Direct causal depthwise conv + downsample-by-2 + gate:
//   y[b,j,d] = xr[b,j,d] * sum_{k=0}^{2j} hw[k,d] * zn[b,2j-k,d]
// Block: 16 j's x 64 d's, 256 threads, each thread 4 j's for one d.
// zn tile (64x64) and hw tile (96x64, zero-filled OOB => causality) in LDS.
// ---------------------------------------------------------------------------
__global__ __launch_bounds__(256) void conv_gate_k(
    const float* __restrict__ zn, const float* __restrict__ hw,
    const float* __restrict__ xr, float* __restrict__ y)
{
  __shared__ float s_z[64 * 64];
  __shared__ float s_h[96 * 64];
  const int tid = threadIdx.x;
  const int j0  = blockIdx.x * 16;
  const int d0  = blockIdx.y * 64;
  const int b   = blockIdx.z;
  const int dl  = tid & 63;
  const int jg  = tid >> 6;                 // 0..3
  float acc[4] = {0.f, 0.f, 0.f, 0.f};
  const int imax = 2 * (j0 + 15);           // causal bound (<= 2046)

  for (int i0 = 0; i0 <= imax; i0 += 64) {
    // zn tile 64x64 (i max = 2047, always in-bounds)
#pragma unroll
    for (int s = 0; s < 4; ++s) {
      int q = tid + 256 * s;                // 0..1023
      int ir = q >> 4, nq = q & 15;
      float4 v = *reinterpret_cast<const float4*>(
          &zn[(((size_t)b * L_IN_) + (i0 + ir)) * DMODEL + d0 + nq * 4]);
      *reinterpret_cast<float4*>(&s_z[ir * 64 + nq * 4]) = v;
    }
    // hw tile 96x64, f = fbase+fr, zero outside [0, L_IN)
    const int fbase = 2 * j0 - i0 - 63;
#pragma unroll
    for (int s = 0; s < 6; ++s) {
      int q = tid + 256 * s;                // 0..1535
      int fr = q >> 4, nq = q & 15;
      int f = fbase + fr;
      float4 v = make_float4(0.f, 0.f, 0.f, 0.f);
      if (f >= 0 && f < L_IN_)
        v = *reinterpret_cast<const float4*>(&hw[(size_t)f * DMODEL + d0 + nq * 4]);
      *reinterpret_cast<float4*>(&s_h[fr * 64 + nq * 4]) = v;
    }
    __syncthreads();

    for (int ii = 0; ii < 64; ++ii) {
      float zv = s_z[ii * 64 + dl];
#pragma unroll
      for (int r = 0; r < 4; ++r) {
        int jl = jg * 4 + r;
        acc[r] += zv * s_h[(2 * jl + 63 - ii) * 64 + dl];
      }
    }
    __syncthreads();
  }

#pragma unroll
  for (int r = 0; r < 4; ++r) {
    int j = j0 + jg * 4 + r;
    size_t idx = (((size_t)b * L_OUT_) + j) * DMODEL + d0 + dl;
    y[idx] = acc[r] * xr[idx];
  }
}

// ---------------------------------------------------------------------------
static void launch_gemm(const float* A, const float* W, const float* bias,
                        const float* res, float* out, int M, int N, int K,
                        bool silu, hipStream_t s)
{
  dim3 grid(N / 64, M / 128), block(256);
  if (silu) {
    gemm_bf16x3_k<true, false><<<grid, block, 0, s>>>(A, W, bias, nullptr, out, M, N, K);
  } else if (res) {
    gemm_bf16x3_k<false, true><<<grid, block, 0, s>>>(A, W, bias, res, out, M, N, K);
  } else {
    gemm_bf16x3_k<false, false><<<grid, block, 0, s>>>(A, W, bias, nullptr, out, M, N, K);
  }
}

extern "C" void kernel_launch(void* const* d_in, const int* in_sizes, int n_in,
                              void* d_out, int out_size, void* d_ws, size_t ws_size,
                              hipStream_t stream)
{
  (void)in_sizes; (void)n_in; (void)out_size; (void)ws_size;
  const float* z       = (const float*)d_in[0];
  const float* x       = (const float*)d_in[1];
  const float* pe      = (const float*)d_in[2];
  const float* W_in    = (const float*)d_in[3];
  const float* b_in    = (const float*)d_in[4];
  const float* g_ln    = (const float*)d_in[5];
  const float* beta_ln = (const float*)d_in[6];
  const float* Wp1     = (const float*)d_in[7];
  const float* bp1     = (const float*)d_in[8];
  const float* Wp2     = (const float*)d_in[9];
  const float* bp2     = (const float*)d_in[10];
  const float* a       = (const float*)d_in[11];
  const float* g1      = (const float*)d_in[12];
  const float* be1     = (const float*)d_in[13];
  const float* W1a     = (const float*)d_in[14];
  const float* b1a     = (const float*)d_in[15];
  const float* W1b     = (const float*)d_in[16];
  const float* b1b     = (const float*)d_in[17];
  const float* g2      = (const float*)d_in[18];
  const float* be2     = (const float*)d_in[19];
  const float* W2a     = (const float*)d_in[20];
  const float* b2a     = (const float*)d_in[21];
  const float* W2b     = (const float*)d_in[22];
  const float* b2b     = (const float*)d_in[23];
  float* out = (float*)d_out;

  // Workspace layout (floats), lifetime-overlapped, total 30M floats = 120 MB.
  float* ws  = (float*)d_ws;
  float* z1  = ws;                              //  8M : z@W_in, then LN in-place
  float* h   = ws + (size_t)8 * 1024 * 1024;    //  8M..10M : filter
  float* xln = ws + (size_t)10 * 1024 * 1024;   // 10M..14M : LN out (t1 overlays)
  float* t1  = xln;                             // 2.62M, dead before xln written
  float* mid = ws + (size_t)14 * 1024 * 1024;   // 14M..22M : FFN hidden
  float* xr  = ws + (size_t)22 * 1024 * 1024;   // 22M..26M : FFN1 out
  float* y1  = ws + (size_t)26 * 1024 * 1024;   // 26M..30M : gated conv out

  const int Mz = BATCH_ * L_IN_;                // 8192
  const int Mx = BATCH_ * L_OUT_;               // 4096

  // 1) z1 = z @ W_in + b_in
  launch_gemm(z, W_in, b_in, nullptr, z1, Mz, DMODEL, DMODEL, false, stream);
  // 2) z1 = LN(z1)
  layernorm_k<<<dim3(Mz), dim3(256), 0, stream>>>(z1, z1, g_ln, beta_ln);
  // 3) t1 = silu(pe @ Wp1 + bp1)
  launch_gemm(pe, Wp1, bp1, nullptr, t1, L_IN_, 1280, 256, true, stream);
  // 4) h = t1 @ Wp2 + bp2
  launch_gemm(t1, Wp2, bp2, nullptr, h, L_IN_, DMODEL, 1280, false, stream);
  // 5) h *= window
  window_k<<<dim3((L_IN_ * DMODEL) / 256), dim3(256), 0, stream>>>(h, a);
  // 6) xln = LN(x)
  layernorm_k<<<dim3(Mx), dim3(256), 0, stream>>>(x, xln, g1, be1);
  // 7) mid = silu(xln @ W1a + b1a)
  launch_gemm(xln, W1a, b1a, nullptr, mid, Mx, 2 * DMODEL, DMODEL, true, stream);
  // 8) xr = mid @ W1b + b1b + x
  launch_gemm(mid, W1b, b1b, x, xr, Mx, DMODEL, 2 * DMODEL, false, stream);
  // 9) y1 = xr * causal_conv(z1, h) downsampled by 2
  conv_gate_k<<<dim3(L_OUT_ / 16, DMODEL / 64, BATCH_), dim3(256), 0, stream>>>(
      z1, h, xr, y1);
  // 10) xln = LN(y1)
  layernorm_k<<<dim3(Mx), dim3(256), 0, stream>>>(y1, xln, g2, be2);
  // 11) mid = silu(xln @ W2a + b2a)
  launch_gemm(xln, W2a, b2a, nullptr, mid, Mx, 2 * DMODEL, DMODEL, true, stream);
  // 12) out = mid @ W2b + b2b + y1
  launch_gemm(mid, W2b, b2b, y1, out, Mx, DMODEL, 2 * DMODEL, false, stream);
}
